// TorchNetPoissonProcess_36000415875730
// MI455X (gfx1250) — matrix operand order repaired and checked by reading, not compiled, not run on hardware
//
#include <hip/hip_runtime.h>
#include <hip/hip_bf16.h>

// Problem constants (from reference: D=20, K=256, N=8192, tau=D, t=N)
#define DW     20
#define KCH    256
#define NOBS   8192
#define T_OUT  8172              // t - tau
#define RED    (DW * KCH)        // 5120 reduction depth
#define BM     64                // block tile rows (time)
#define BN     128               // block tile cols (out channels)
#define LDSP   264               // padded LDS row stride in f16 (breaks bank conflicts)
#define AROWS  (BM + DW - 1)     // 83 obs rows staged per block

typedef __attribute__((ext_vector_type(16))) _Float16 v16h;
typedef __attribute__((ext_vector_type(8)))  _Float16 v8h;
typedef __attribute__((ext_vector_type(4)))  _Float16 v4h;
typedef __attribute__((ext_vector_type(8)))  float    v8f;

// ---------------------------------------------------------------------------
// Prep: Bt[n][kk] = (f16) beta1[n, b, a]   with kk = a*256 + b
// Makes each lane's WMMA B-fragment two contiguous 16-byte reads.
// ---------------------------------------------------------------------------
__global__ __launch_bounds__(256) void pack_beta1_kernel(
    const float* __restrict__ beta1, _Float16* __restrict__ Bt)
{
    int idx = blockIdx.x * 256 + threadIdx.x;       // 0 .. 5120*256-1
    int n  = idx / RED;
    int kk = idx - n * RED;
    int a  = kk >> 8;
    int b  = kk & 255;
    Bt[(size_t)n * RED + kk] = (_Float16)beta1[(size_t)n * RED + b * DW + a];
}

// ---------------------------------------------------------------------------
// Main: sliding-window GEMM via v_wmma_f32_16x16x32_f16 + loglik partials
// ---------------------------------------------------------------------------
__global__ __launch_bounds__(256) void poisson_gemm_kernel(
    const float* __restrict__ obs,     // [8192, 256] f32
    const float* __restrict__ beta0,   // [256] f32
    const _Float16* __restrict__ Bt,   // [256, 5120] f16 (transposed beta1)
    float* __restrict__ out,           // out[0]=loglik, out[1..] = lam [T,K]
    float* __restrict__ partials)      // per-block loglik partials
{
    __shared__ alignas(16) _Float16 As[AROWS * LDSP];
    __shared__ float wsum[8];

    const int tid  = threadIdx.x;
    const int lane = tid & 31;
    const int wave = tid >> 5;
    const int hv   = lane >> 4;        // half of wave (0/1)
    const int ln   = lane & 15;
    const int wm   = wave & 1;         // wave M index (0..1), 32 rows each
    const int wn   = wave >> 1;        // wave N index (0..3), 32 cols each
    const int i0   = blockIdx.x * BM;  // first output row of block
    const int nb0  = blockIdx.y * BN;  // first output channel of block

    // ---- Stage obs rows [i0, i0+83) into LDS as f16 (vectorized x4) ----
    for (int idx = tid; idx < AROWS * (KCH / 4); idx += 256) {
        int r  = idx >> 6;             // / 64
        int c4 = (idx & 63) << 2;      // column *4
        int row = i0 + r;
        if (row > NOBS - 1) row = NOBS - 1;   // clamped rows feed discarded outputs
        const float4 v = *(const float4*)&obs[(size_t)row * KCH + c4];
        v4h h;
        h[0] = (_Float16)v.x; h[1] = (_Float16)v.y;
        h[2] = (_Float16)v.z; h[3] = (_Float16)v.w;
        *(v4h*)&As[r * LDSP + c4] = h;
    }
    __syncthreads();

    v8f acc[2][2] = {};
    const int rowL0 = wm * 32 + ln;    // LDS-local row for A fragment (fm adds 16)

    // ---- K loop: 160 iterations of depth-32 WMMA ----
    for (int K0 = 0; K0 < RED; K0 += 32) {
        // A fragment per ISA 16-bit 16x32 layout:
        //   slot s<8  -> K = hv*8 + s        (run0: 8 contiguous kk)
        //   slot s>=8 -> K = 16 + hv*8 + s-8 (run1: 8 contiguous kk)
        const int kk0 = K0 + hv * 8;
        const int kk1 = K0 + 16 + hv * 8;
        const int a0 = kk0 >> 8, b0 = kk0 & 255;
        const int a1 = kk1 >> 8, b1 = kk1 & 255;

        v16h afrag[2];
#pragma unroll
        for (int fm = 0; fm < 2; ++fm) {
            const int rl = rowL0 + fm * 16;
            const v8h lo = *(const v8h*)&As[(rl + a0) * LDSP + b0];
            const v8h hi = *(const v8h*)&As[(rl + a1) * LDSP + b1];
            v16h f;
#pragma unroll
            for (int s = 0; s < 8; ++s) { f[s] = lo[s]; f[s + 8] = hi[s]; }
            afrag[fm] = f;
        }

        // B fragment: same K order as A (s<8 -> hv*8+s, s>=8 -> 16+hv*8+s-8), N = ln
        v16h bfrag[2];
#pragma unroll
        for (int fn = 0; fn < 2; ++fn) {
            const int n = nb0 + wn * 32 + fn * 16 + ln;
            const v8h blo = *(const v8h*)&Bt[(size_t)n * RED + K0 + hv * 8];
            const v8h bhi = *(const v8h*)&Bt[(size_t)n * RED + K0 + 16 + hv * 8];
            v16h g;
#pragma unroll
            for (int s = 0; s < 8; ++s) { g[s] = blo[s]; g[s + 8] = bhi[s]; }
            bfrag[fn] = g;
        }

#pragma unroll
        for (int fm = 0; fm < 2; ++fm)
#pragma unroll
            for (int fn = 0; fn < 2; ++fn)
                acc[fm][fn] = __builtin_amdgcn_wmma_f32_16x16x32_f16(
                    false, afrag[fm], false, bfrag[fn],
                    (short)0, acc[fm][fn], false, false);
    }

    // ---- Epilogue: lam = acc + beta0; store; accumulate loglik partial ----
    float part = 0.f;
#pragma unroll
    for (int fm = 0; fm < 2; ++fm) {
#pragma unroll
        for (int fn = 0; fn < 2; ++fn) {
            const int k = nb0 + wn * 32 + fn * 16 + ln;
            const float bv = beta0[k];
#pragma unroll
            for (int r = 0; r < 8; ++r) {
                // D layout: VGPR r -> M = r + 8*hv, N = ln
                const int i = i0 + wm * 32 + fm * 16 + r + 8 * hv;
                const float lam = acc[fm][fn][r] + bv;
                if (i < T_OUT) {
                    out[1 + (size_t)i * KCH + k] = lam;
                    const float o = obs[(size_t)(i + DW) * KCH + k]; // obs[tau + i]
                    part += o * __logf(lam) - lam;
                }
            }
        }
    }

    // fixed-order wave reduction (wave32)
#pragma unroll
    for (int off = 16; off > 0; off >>= 1)
        part += __shfl_xor(part, off, 32);
    if (lane == 0) wsum[wave] = part;
    __syncthreads();
    if (tid == 0) {
        float s = 0.f;
#pragma unroll
        for (int w = 0; w < 8; ++w) s += wsum[w];
        partials[blockIdx.y * gridDim.x + blockIdx.x] = s;
    }
}

// ---------------------------------------------------------------------------
// Deterministic final reduction of per-block partials -> out[0]
// ---------------------------------------------------------------------------
__global__ void finalize_kernel(const float* __restrict__ partials, int n,
                                float* __restrict__ out)
{
    if (threadIdx.x == 0 && blockIdx.x == 0) {
        double s = 0.0;
        for (int i = 0; i < n; ++i) s += (double)partials[i];
        out[0] = (float)s;
    }
}

extern "C" void kernel_launch(void* const* d_in, const int* in_sizes, int n_in,
                              void* d_out, int out_size, void* d_ws, size_t ws_size,
                              hipStream_t stream) {
    (void)in_sizes; (void)n_in; (void)out_size; (void)ws_size;
    const float* obs   = (const float*)d_in[0];   // [8192,256]
    const float* beta0 = (const float*)d_in[1];   // [1,256]
    const float* beta1 = (const float*)d_in[2];   // [256,256,20]
    // d_in[3] = tau (=20), d_in[4] = t (=8192): compile-time constants here.

    _Float16* Bt       = (_Float16*)d_ws;                               // 2.62 MB
    float*    partials = (float*)((char*)d_ws + (size_t)KCH * RED * 2); // 256 floats
    float*    out      = (float*)d_out;

    // 1) transpose/convert beta1 -> Bt (f16)
    pack_beta1_kernel<<<(KCH * RED) / 256, 256, 0, stream>>>(beta1, Bt);

    // 2) WMMA GEMM + partial loglik
    dim3 grid((T_OUT + BM - 1) / BM, KCH / BN);   // 128 x 2
    poisson_gemm_kernel<<<grid, 256, 0, stream>>>(obs, beta0, Bt, out, partials);

    // 3) deterministic scalar reduction
    finalize_kernel<<<1, 32, 0, stream>>>(partials, grid.x * grid.y, out);
}
